// MultiHeadAttention_45234595561648
// MI455X (gfx1250) — compile-verified, matrix operand order checked by reading, not run on hardware
//
#include <hip/hip_runtime.h>
#include <hip/hip_bf16.h>

// ---------------------------------------------------------------------------
// MI455X (gfx1250) multi-head causal attention, bf16 WMMA path.
//   x[4096,1024] @ Wq/Wk/Wv -> Q,K,V (bf16)
//   per-head flash attention with online softmax (f32 accum)
//   attn @ Wo -> out (fp32)
// All matmuls use v_wmma_f32_16x16x32_bf16 (wave32); next-tile
// global_prefetch_b8 hides LDS staging behind WMMA issue.
// ---------------------------------------------------------------------------

typedef __bf16 bf16;
typedef __attribute__((ext_vector_type(16))) __bf16 bf16x16;
typedef __attribute__((ext_vector_type(8)))  __bf16 bf16x8;
typedef __attribute__((ext_vector_type(4)))  __bf16 bf16x4;
typedef __attribute__((ext_vector_type(8)))  float  f32x8;

#define SEQ    4096
#define DMODEL 1024
#define NHEADS 16
#define HDIM   64

static __device__ __forceinline__ bf16 cvt_bf16(float f) {
  union { float f; unsigned u; } v; v.f = f;
  unsigned r = (v.u + 0x7FFFu + ((v.u >> 16) & 1u)) >> 16;   // RNE
  union { unsigned short u; bf16 b; } o; o.u = (unsigned short)r;
  return o.b;
}

static __device__ __forceinline__ f32x8 wmma_bf16(bf16x16 a, bf16x16 b, f32x8 c) {
  // emits v_wmma_f32_16x16x32_bf16
  return __builtin_amdgcn_wmma_f32_16x16x32_bf16(
      /*neg_a=*/false, a, /*neg_b=*/false, b,
      /*c_mod=*/(short)0, c, /*reuse_a=*/false, /*reuse_b=*/false);
}

// Load a 16x32 bf16 A-style fragment (also valid for B = rows of the
// transposed operand) from an LDS tile stored row-major with `pitch` bf16.
// ISA layout: lane-half g holds K = {8g..8g+7, 16+8g..16+8g+7}, row = lane&15.
static __device__ __forceinline__ bf16x16 lds_frag(const bf16* base, int pitch) {
  const int lane = threadIdx.x & 31;
  const int g = lane >> 4;
  const int r = lane & 15;
  const bf16* p = base + r * pitch + g * 8;
  bf16x8 lo = *(const bf16x8*)(p);        // ds_load_b128
  bf16x8 hi = *(const bf16x8*)(p + 16);   // ds_load_b128
  bf16x16 out;
#pragma unroll
  for (int i = 0; i < 8; ++i) { out[i] = lo[i]; out[i + 8] = hi[i]; }
  return out;
}

// ---------------------------------------------------------------------------
// fp32 -> bf16 conversion (n multiple of 4)
// ---------------------------------------------------------------------------
__global__ __launch_bounds__(256)
void cvt_f32_bf16_kernel(const float* __restrict__ in, bf16* __restrict__ out, int n) {
  int i = (blockIdx.x * blockDim.x + threadIdx.x) * 4;
  if (i >= n) return;
  float4 v = *(const float4*)(in + i);
  bf16x4 o;
  o[0] = cvt_bf16(v.x); o[1] = cvt_bf16(v.y);
  o[2] = cvt_bf16(v.z); o[3] = cvt_bf16(v.w);
  *(bf16x4*)(out + i) = o;
}

// ---------------------------------------------------------------------------
// C[M,N] = A[M,K] @ B[K,N], bf16 in, f32 accumulate.
// 128 threads = 4 waves; workgroup tile 128x64; wave w owns rows 32w..32w+31
// (two 16-row sub-tiles). K staged 64 wide -> 16 WMMAs per barrier pair.
// ---------------------------------------------------------------------------
template <bool OUT_BF16>
__global__ __launch_bounds__(128)
void gemm_bf16_kernel(const bf16* __restrict__ A, const bf16* __restrict__ B,
                      void* __restrict__ Cout, int M, int N, int K) {
  __shared__ __align__(16) bf16 As[128 * 72];  // A tile 128x64, pitch 72
  __shared__ __align__(16) bf16 Bt[64 * 72];   // B^T tile: Bt[n][k 0..63]

  const int tid  = threadIdx.x;
  const int wave = tid >> 5;
  const int lane = tid & 31;
  const int g = lane >> 4, ln = lane & 15;
  const int m0 = blockIdx.x * 128;
  const int n0 = blockIdx.y * 64;

  f32x8 acc[2][4];
#pragma unroll
  for (int mi = 0; mi < 2; ++mi)
#pragma unroll
    for (int t = 0; t < 4; ++t)
#pragma unroll
      for (int j = 0; j < 8; ++j) acc[mi][t][j] = 0.0f;

  for (int kk = 0; kk < K; kk += 64) {
    __syncthreads();
    // stage A rows: 128 rows x 64 k, b128 loads
#pragma unroll
    for (int i = 0; i < 8; ++i) {
      int L = tid + i * 128;            // 0..1023
      int r = L >> 3;                   // 0..127
      int d = (L & 7) * 8;              // 0..56
      bf16x8 v = *(const bf16x8*)(A + (size_t)(m0 + r) * K + kk + d);
      *(bf16x8*)(As + r * 72 + d) = v;
      if (kk + 64 < K)                  // prefetch next A chunk into L2/L0
        __builtin_prefetch(A + (size_t)(m0 + r) * K + kk + 64 + d, 0, 1);
    }
    // stage B^T: 64 k x 64 n (coalesced along n, scatter bf16 into Bt[n][k])
#pragma unroll
    for (int i = 0; i < 8; ++i) {
      int L = tid + i * 128;            // 0..1023
      int k = L >> 4;                   // 0..63
      int n = (L & 15) * 4;             // 0..60
      bf16x4 v = *(const bf16x4*)(B + (size_t)(kk + k) * N + n0 + n);
#pragma unroll
      for (int j = 0; j < 4; ++j) Bt[(n + j) * 72 + k] = v[j];
      if (kk + 64 < K)
        __builtin_prefetch(B + (size_t)(kk + 64 + k) * N + n0 + n, 0, 1);
    }
    __syncthreads();

#pragma unroll
    for (int kc = 0; kc < 2; ++kc) {
      bf16x16 a0 = lds_frag(As + (wave * 32 +  0) * 72 + kc * 32, 72);
      bf16x16 a1 = lds_frag(As + (wave * 32 + 16) * 72 + kc * 32, 72);
#pragma unroll
      for (int t = 0; t < 4; ++t) {
        bf16x16 b = lds_frag(Bt + t * 16 * 72 + kc * 32, 72);
        acc[0][t] = wmma_bf16(a0, b, acc[0][t]);
        acc[1][t] = wmma_bf16(a1, b, acc[1][t]);
      }
    }
  }

  // C/D layout: VGPR j -> row (j + 8g), col = ln within 16-wide tile
#pragma unroll
  for (int mi = 0; mi < 2; ++mi)
#pragma unroll
    for (int t = 0; t < 4; ++t)
#pragma unroll
      for (int j = 0; j < 8; ++j) {
        int row = m0 + wave * 32 + mi * 16 + g * 8 + j;
        int col = n0 + t * 16 + ln;
        float v = acc[mi][t][j];
        if (OUT_BF16) ((bf16*)Cout)[(size_t)row * N + col] = cvt_bf16(v);
        else          ((float*)Cout)[(size_t)row * N + col] = v;
      }
}

// ---------------------------------------------------------------------------
// Causal flash attention, one workgroup = (64 queries x 1 head), 4 waves.
// Q,K,V,O are bf16 [SEQ, DMODEL] with head h occupying cols h*64..h*64+63.
// ---------------------------------------------------------------------------
__global__ __launch_bounds__(128)
void attn_kernel(const bf16* __restrict__ Q, const bf16* __restrict__ Kb,
                 const bf16* __restrict__ Vb, bf16* __restrict__ Ob) {
  __shared__ __align__(16) bf16 Ks[64 * 72];       // K tile row-major [key][d]
  __shared__ __align__(16) bf16 Vt[64 * 72];       // V^T tile [d][key]
  __shared__ __align__(16) bf16 Ps[4 * 16 * 72];   // per-wave P tile [16][64]

  const int tid  = threadIdx.x;
  const int wave = tid >> 5;
  const int lane = tid & 31;
  const int g = lane >> 4, ln = lane & 15;
  const int q0   = blockIdx.x * 64;
  const int hoff = blockIdx.y * HDIM;

  // Q fragments (16 rows x 64 dims) loaded straight from global in A-layout
  bf16x16 qf[2];
  {
    const bf16* qp = Q + (size_t)(q0 + wave * 16 + ln) * DMODEL + hoff + g * 8;
#pragma unroll
    for (int c = 0; c < 2; ++c) {
      bf16x8 lo = *(const bf16x8*)(qp + c * 32);
      bf16x8 hi = *(const bf16x8*)(qp + c * 32 + 16);
#pragma unroll
      for (int i = 0; i < 8; ++i) { qf[c][i] = lo[i]; qf[c][i + 8] = hi[i]; }
    }
  }

  f32x8 o[4];
  float m[8], l[8];
#pragma unroll
  for (int t = 0; t < 4; ++t)
#pragma unroll
    for (int j = 0; j < 8; ++j) o[t][j] = 0.0f;
#pragma unroll
  for (int j = 0; j < 8; ++j) { m[j] = -1e30f; l[j] = 0.0f; }

  const int nkb = q0 / 64 + 1;   // causal: key blocks 0..q0/64
  for (int kb = 0; kb < nkb; ++kb) {
    const int k0 = kb * 64;
    __syncthreads();   // previous iteration's LDS reads done
    // stage K tile rows (b128 loads) + prefetch next block
#pragma unroll
    for (int i = 0; i < 4; ++i) {
      int L = tid + i * 128;            // 0..511
      int s = L >> 3;                   // key 0..63
      int d = (L & 7) * 8;              // 0..56
      bf16x8 v = *(const bf16x8*)(Kb + (size_t)(k0 + s) * DMODEL + hoff + d);
      *(bf16x8*)(Ks + s * 72 + d) = v;
      if (kb + 1 < nkb) {
        __builtin_prefetch(Kb + (size_t)(k0 + 64 + s) * DMODEL + hoff + d, 0, 1);
        __builtin_prefetch(Vb + (size_t)(k0 + 64 + s) * DMODEL + hoff + d, 0, 1);
      }
    }
    // stage V^T (coalesced read, bf16 scatter)
#pragma unroll
    for (int i = 0; i < 8; ++i) {
      int L = tid + i * 128;            // 0..1023
      int s = L >> 4;                   // key 0..63
      int d = (L & 15) * 4;             // 0..60
      bf16x4 v = *(const bf16x4*)(Vb + (size_t)(k0 + s) * DMODEL + hoff + d);
#pragma unroll
      for (int j = 0; j < 4; ++j) Vt[(d + j) * 72 + s] = v[j];
    }
    __syncthreads();

    // S = Q K^T  (4 key tiles x 2 K-dim chunks = 8 WMMAs)
    f32x8 sc[4];
#pragma unroll
    for (int t = 0; t < 4; ++t) {
      f32x8 z = {0.f, 0.f, 0.f, 0.f, 0.f, 0.f, 0.f, 0.f};
      sc[t] = z;
#pragma unroll
      for (int c = 0; c < 2; ++c) {
        bf16x16 kf = lds_frag(Ks + t * 16 * 72 + c * 32, 72);
        sc[t] = wmma_bf16(qf[c], kf, sc[t]);
      }
    }

    // scale + causal mask (diagonal block only)
    const bool diag = (k0 == q0);
#pragma unroll
    for (int t = 0; t < 4; ++t)
#pragma unroll
      for (int j = 0; j < 8; ++j) {
        float v = sc[t][j] * 0.125f;     // 1/sqrt(64)
        if (diag) {
          int kg = k0 + t * 16 + ln;
          int qg = q0 + wave * 16 + g * 8 + j;
          if (kg > qg) v = -1e30f;
        }
        sc[t][j] = v;
      }

    // online softmax: row j+8g lives at VGPR index j, spread over 16 lanes
#pragma unroll
    for (int j = 0; j < 8; ++j) {
      float bm = fmaxf(fmaxf(sc[0][j], sc[1][j]), fmaxf(sc[2][j], sc[3][j]));
#pragma unroll
      for (int msk = 1; msk < 16; msk <<= 1)
        bm = fmaxf(bm, __shfl_xor(bm, msk, 32));
      float mn = fmaxf(m[j], bm);
      float rescale = __expf(m[j] - mn);
      l[j] *= rescale;
#pragma unroll
      for (int t = 0; t < 4; ++t) o[t][j] *= rescale;
      float rsum = 0.0f;
#pragma unroll
      for (int t = 0; t < 4; ++t) {
        float p = __expf(sc[t][j] - mn);
        sc[t][j] = p;
        rsum += p;
      }
#pragma unroll
      for (int msk = 1; msk < 16; msk <<= 1)
        rsum += __shfl_xor(rsum, msk, 32);
      l[j] += rsum;
      m[j] = mn;
    }

    // re-fragment P: C-layout -> LDS -> A-layout
    bf16* pw = Ps + wave * 16 * 72;
#pragma unroll
    for (int t = 0; t < 4; ++t)
#pragma unroll
      for (int j = 0; j < 8; ++j)
        pw[(g * 8 + j) * 72 + t * 16 + ln] = cvt_bf16(sc[t][j]);
    __syncthreads();   // order P stores before fragment reloads

    // O += P V  (2 chunks x 4 feature tiles = 8 WMMAs)
#pragma unroll
    for (int c = 0; c < 2; ++c) {
      bf16x16 pf = lds_frag(pw + c * 32, 72);
#pragma unroll
      for (int t = 0; t < 4; ++t) {
        bf16x16 vf = lds_frag(Vt + t * 16 * 72 + c * 32, 72);
        o[t] = wmma_bf16(pf, vf, o[t]);
      }
    }
  }

  // normalize and write bf16 attention output
#pragma unroll
  for (int j = 0; j < 8; ++j) l[j] = 1.0f / l[j];
#pragma unroll
  for (int t = 0; t < 4; ++t)
#pragma unroll
    for (int j = 0; j < 8; ++j) {
      int row = q0 + wave * 16 + g * 8 + j;
      int col = hoff + t * 16 + ln;
      Ob[(size_t)row * DMODEL + col] = cvt_bf16(o[t][j] * l[j]);
    }
}

// ---------------------------------------------------------------------------
extern "C" void kernel_launch(void* const* d_in, const int* in_sizes, int n_in,
                              void* d_out, int out_size, void* d_ws, size_t ws_size,
                              hipStream_t stream) {
  const float* x  = (const float*)d_in[0];
  const float* Wq = (const float*)d_in[1];
  const float* Wk = (const float*)d_in[2];
  const float* Wv = (const float*)d_in[3];
  const float* Wo = (const float*)d_in[4];

  const size_t NX = (size_t)SEQ * DMODEL;      // 4,194,304
  const size_t NW = (size_t)DMODEL * DMODEL;   // 1,048,576

  char* ws = (char*)d_ws;
  bf16* xb  = (bf16*)(ws);                     size_t off = NX * 2;
  bf16* Wqb = (bf16*)(ws + off);               off += NW * 2;
  bf16* Wkb = (bf16*)(ws + off);               off += NW * 2;
  bf16* Wvb = (bf16*)(ws + off);               off += NW * 2;
  bf16* Wob = (bf16*)(ws + off);               off += NW * 2;
  bf16* Qb  = (bf16*)(ws + off);               off += NX * 2;
  bf16* Kb  = (bf16*)(ws + off);               off += NX * 2;
  bf16* Vb  = (bf16*)(ws + off);               off += NX * 2;
  bf16* Ab  = (bf16*)(ws + off);               off += NX * 2;   // attention out

  // fp32 -> bf16 conversions
  cvt_f32_bf16_kernel<<<(int)(NX / 4 / 256), 256, 0, stream>>>(x,  xb,  (int)NX);
  cvt_f32_bf16_kernel<<<(int)(NW / 4 / 256), 256, 0, stream>>>(Wq, Wqb, (int)NW);
  cvt_f32_bf16_kernel<<<(int)(NW / 4 / 256), 256, 0, stream>>>(Wk, Wkb, (int)NW);
  cvt_f32_bf16_kernel<<<(int)(NW / 4 / 256), 256, 0, stream>>>(Wv, Wvb, (int)NW);
  cvt_f32_bf16_kernel<<<(int)(NW / 4 / 256), 256, 0, stream>>>(Wo, Wob, (int)NW);

  dim3 ggrid(SEQ / 128, DMODEL / 64);
  gemm_bf16_kernel<true><<<ggrid, 128, 0, stream>>>(xb, Wqb, (void*)Qb, SEQ, DMODEL, DMODEL);
  gemm_bf16_kernel<true><<<ggrid, 128, 0, stream>>>(xb, Wkb, (void*)Kb, SEQ, DMODEL, DMODEL);
  gemm_bf16_kernel<true><<<ggrid, 128, 0, stream>>>(xb, Wvb, (void*)Vb, SEQ, DMODEL, DMODEL);

  dim3 agrid(SEQ / 64, NHEADS);
  attn_kernel<<<agrid, 128, 0, stream>>>(Qb, Kb, Vb, Ab);

  gemm_bf16_kernel<false><<<ggrid, 128, 0, stream>>>(Ab, Wob, d_out, SEQ, DMODEL, DMODEL);
}
